// Decoder_7541962571852
// MI455X (gfx1250) — matrix operand order repaired and checked by reading, not compiled, not run on hardware
//
#include <hip/hip_runtime.h>

#define LNUM 2
#define HDIM 128
#define BDIM 8192
#define OBS_T 8
#define PRED_T 30

typedef _Float16 half_t;
typedef __attribute__((ext_vector_type(16))) _Float16 v16h;
typedef __attribute__((ext_vector_type(8)))  _Float16 v8h;
typedef __attribute__((ext_vector_type(8)))  float    v8f;

__device__ __forceinline__ float sigmoidf(float x) { return 1.0f / (1.0f + __expf(-x)); }

// Load one 32x16 B-operand fragment (fp16) from a row-major f32 weight matrix
// W[n][k].  B order used here: lane holds column n = tilebase+(lane&15),
// halves 0..15 = K (lane>>4)*16 .. +15 contiguous.
__device__ __forceinline__ v16h load_wfrag(const float* __restrict__ W, int kdim, int n, int k0) {
  const float* p = W + (size_t)n * kdim + k0;
  v16h out;
#pragma unroll
  for (int i = 0; i < 16; ++i) out[i] = (half_t)p[i];
  return out;
}

// Load one 16x32 A-operand fragment from row-major fp16 LDS (row = lane&15).
// Contiguous A order (matches B): halves 0..7 = K base0..base0+7, halves 8..15 =
// K base0+8..+15, base0 = (lane>>4)*16.  Two 16B LDS reads.
__device__ __forceinline__ v16h load_afrag(const half_t* p_row, int base0, int kchunk) {
  const v8h lo = *(const v8h*)(p_row + kchunk * 32 + base0);
  const v8h hi = *(const v8h*)(p_row + kchunk * 32 + base0 + 8);
  v16h a;
#pragma unroll
  for (int i = 0; i < 8; ++i) { a[i] = lo[i]; a[i + 8] = hi[i]; }
  return a;
}

__global__ __launch_bounds__(256, 1) void decoder_lstm_kernel(
    const float* __restrict__ obs,  const float* __restrict__ h0g, const float* __restrict__ c0g,
    const float* __restrict__ Wih0, const float* __restrict__ Whh0,
    const float* __restrict__ bih0, const float* __restrict__ bhh0,
    const float* __restrict__ Wih1, const float* __restrict__ Whh1,
    const float* __restrict__ bih1, const float* __restrict__ bhh1,
    const float* __restrict__ Whp,  const float* __restrict__ bhp,
    float* __restrict__ out)
{
  __shared__ __align__(16) float  lds_gates[16 * 512];   // 32 KB
  __shared__ __align__(16) half_t lds_h0[16 * HDIM];     // 4 KB
  __shared__ __align__(16) half_t lds_h1[16 * HDIM];     // 4 KB
  __shared__ __align__(16) float  lds_xf[16 * 4];        // current x (3 used)
  __shared__ __align__(16) float  lds_whp[3 * HDIM];     // 1.5 KB
  __shared__ __align__(16) float  lds_econ[128 * 20];    // 10 KB: per-column
                                    // [0:4)=b0(i,f,g,o) [4:8)=b1 [8:20)=Wih0 rows

  const int tid  = threadIdx.x;
  const int w    = tid >> 5;       // wave id 0..7, owns gate cols [64w, 64w+64)
  const int lane = tid & 31;
  const int ln15 = lane & 15;
  const int lhi  = lane >> 4;
  const int rowbase = blockIdx.x * 16;

  // ---- stage initial state into LDS ----
  for (int i = tid; i < 3 * HDIM; i += 256) lds_whp[i] = Whp[i];
  for (int i = tid; i < 16 * HDIM; i += 256) {
    int r = i >> 7, c = i & 127;
    size_t g = (size_t)(rowbase + r) * HDIM + c;
    lds_h0[i] = (half_t)h0g[g];
    lds_h1[i] = (half_t)h0g[(size_t)BDIM * HDIM + g];
  }
  if (tid < 16) {
    int r = tid;
    const float* px = obs + (size_t)(OBS_T - 1) * BDIM * 3 + (size_t)(rowbase + r) * 3;
    lds_xf[r * 4 + 0] = px[0];
    lds_xf[r * 4 + 1] = px[1];
    lds_xf[r * 4 + 2] = px[2];
    lds_xf[r * 4 + 3] = 0.0f;
  }
  if (tid < 128) {
    int col = tid;
    float* e = lds_econ + col * 20;
#pragma unroll
    for (int g = 0; g < 4; ++g) {
      int n = col + 128 * g;
      e[g]         = bih0[n] + bhh0[n];
      e[4 + g]     = bih1[n] + bhh1[n];
      e[8 + g * 3 + 0] = Wih0[n * 3 + 0];
      e[8 + g * 3 + 1] = Wih0[n * 3 + 1];
      e[8 + g * 3 + 2] = Wih0[n * 3 + 2];
    }
  }

  // ---- cell state in registers: wave owns h-columns [16w,16w+16) ----
  const int ccol = 16 * w + ln15;
  float c0s[8], c1s[8];
#pragma unroll
  for (int j = 0; j < 8; ++j) {
    int r = 8 * lhi + j;
    size_t g = (size_t)(rowbase + r) * HDIM + ccol;
    c0s[j] = c0g[g];
    c1s[j] = c0g[(size_t)BDIM * HDIM + g];
  }

  // ---- weight fragments resident in VGPRs for the whole kernel ----
  const int kb16 = lhi * 16;
  v16h w0h[4][4];   // Whh0:        K=128 -> 4 k-chunks x 4 n-tiles
  v16h w1f[8][4];   // [Wih1;Whh1]: K=256 -> 8 k-chunks x 4 n-tiles
#pragma unroll
  for (int t = 0; t < 4; ++t) {
    int n = 64 * w + 16 * t + ln15;
#pragma unroll
    for (int kc = 0; kc < 4; ++kc) w0h[kc][t] = load_wfrag(Whh0, HDIM, n, kc * 32 + kb16);
#pragma unroll
    for (int kc = 0; kc < 8; ++kc) {
      const float* src = (kc < 4) ? Wih1 : Whh1;
      w1f[kc][t] = load_wfrag(src, HDIM, n, (kc & 3) * 32 + kb16);
    }
  }

  const int abase0 = lhi * 16;
  const half_t* arow_h0 = lds_h0 + ln15 * HDIM;
  const half_t* arow_h1 = lds_h1 + ln15 * HDIM;
  const float4* econ = (const float4*)(lds_econ + ccol * 20);

  __syncthreads();

  for (int step = 0; step < PRED_T; ++step) {
    // ======== layer 0 matmul: gates0 = h0 @ Whh0^T (2 passes x 2 N-tiles) ========
#pragma unroll
    for (int hf = 0; hf < 2; ++hf) {
      v8f acc[2];
#pragma unroll
      for (int u = 0; u < 2; ++u) acc[u] = (v8f){};
#pragma unroll
      for (int kc = 0; kc < 4; ++kc) {
        v16h a = load_afrag(arow_h0, abase0, kc);
#pragma unroll
        for (int u = 0; u < 2; ++u)
          acc[u] = __builtin_amdgcn_wmma_f32_16x16x32_f16(false, a, false, w0h[kc][2 * hf + u],
                                                          (short)0, acc[u], false, false);
      }
#pragma unroll
      for (int u = 0; u < 2; ++u) {
        int col = 64 * w + 16 * (2 * hf + u) + ln15;
#pragma unroll
        for (int v = 0; v < 8; ++v) lds_gates[(v + 8 * lhi) * 512 + col] = acc[u][v];
      }
    }
    __syncthreads();

    // ======== layer 0 elementwise: + bias + x@Wih0^T, LSTM update ========
    {
      float4 B0 = econ[0];
      float4 W0 = econ[2], W1 = econ[3], W2 = econ[4];
#pragma unroll
      for (int j = 0; j < 8; ++j) {
        int r = 8 * lhi + j;
        float x0 = lds_xf[r * 4 + 0], x1 = lds_xf[r * 4 + 1], x2 = lds_xf[r * 4 + 2];
        float gi = lds_gates[r * 512 + ccol      ] + B0.x + W0.x * x0 + W0.y * x1 + W0.z * x2;
        float gf = lds_gates[r * 512 + ccol + 128] + B0.y + W0.w * x0 + W1.x * x1 + W1.y * x2;
        float gg = lds_gates[r * 512 + ccol + 256] + B0.z + W1.z * x0 + W1.w * x1 + W2.x * x2;
        float go = lds_gates[r * 512 + ccol + 384] + B0.w + W2.y * x0 + W2.z * x1 + W2.w * x2;
        float cn = sigmoidf(gf) * c0s[j] + sigmoidf(gi) * tanhf(gg);
        c0s[j] = cn;
        lds_h0[r * HDIM + ccol] = (half_t)(sigmoidf(go) * tanhf(cn));
      }
    }
    __syncthreads();

    // ======== layer 1 matmul: gates1 = [h0_new;h1_old] @ [Wih1;Whh1]^T ========
#pragma unroll
    for (int hf = 0; hf < 2; ++hf) {
      v8f acc[2];
#pragma unroll
      for (int u = 0; u < 2; ++u) acc[u] = (v8f){};
#pragma unroll
      for (int kc = 0; kc < 8; ++kc) {
        v16h a = (kc < 4) ? load_afrag(arow_h0, abase0, kc)
                          : load_afrag(arow_h1, abase0, kc - 4);
#pragma unroll
        for (int u = 0; u < 2; ++u)
          acc[u] = __builtin_amdgcn_wmma_f32_16x16x32_f16(false, a, false, w1f[kc][2 * hf + u],
                                                          (short)0, acc[u], false, false);
      }
#pragma unroll
      for (int u = 0; u < 2; ++u) {
        int col = 64 * w + 16 * (2 * hf + u) + ln15;
#pragma unroll
        for (int v = 0; v < 8; ++v) lds_gates[(v + 8 * lhi) * 512 + col] = acc[u][v];
      }
    }
    __syncthreads();

    // ======== layer 1 elementwise ========
    {
      float4 B1 = econ[1];
#pragma unroll
      for (int j = 0; j < 8; ++j) {
        int r = 8 * lhi + j;
        float gi = lds_gates[r * 512 + ccol      ] + B1.x;
        float gf = lds_gates[r * 512 + ccol + 128] + B1.y;
        float gg = lds_gates[r * 512 + ccol + 256] + B1.z;
        float go = lds_gates[r * 512 + ccol + 384] + B1.w;
        float cn = sigmoidf(gf) * c1s[j] + sigmoidf(gi) * tanhf(gg);
        c1s[j] = cn;
        lds_h1[r * HDIM + ccol] = (half_t)(sigmoidf(go) * tanhf(cn));
      }
    }
    __syncthreads();

    // ======== hidden2pos: out = h1_new @ Whp^T + bhp ; feeds next x ========
    // 3 waves, split-K over lane halves, combined with one cross-half shuffle.
    if (w < 3) {
      int r = ln15;
      int kb = 64 * lhi;
      float a = 0.0f;
#pragma unroll 8
      for (int k = kb; k < kb + 64; ++k)
        a += (float)lds_h1[r * HDIM + k] * lds_whp[w * HDIM + k];
      a += __shfl_xor(a, 16, 32);
      if (lhi == 0) {
        float o = a + bhp[w];
        out[(size_t)step * BDIM * 3 + (size_t)(rowbase + r) * 3 + w] = o;
        lds_xf[r * 4 + w] = o;
      }
    }
    __syncthreads();
  }
}

extern "C" void kernel_launch(void* const* d_in, const int* in_sizes, int n_in,
                              void* d_out, int out_size, void* d_ws, size_t ws_size,
                              hipStream_t stream) {
  (void)in_sizes; (void)n_in; (void)out_size; (void)d_ws; (void)ws_size;
  const float* obs  = (const float*)d_in[0];
  const float* h0g  = (const float*)d_in[1];
  const float* c0g  = (const float*)d_in[2];
  const float* Wih0 = (const float*)d_in[3];
  const float* Whh0 = (const float*)d_in[4];
  const float* bih0 = (const float*)d_in[5];
  const float* bhh0 = (const float*)d_in[6];
  const float* Wih1 = (const float*)d_in[7];
  const float* Whh1 = (const float*)d_in[8];
  const float* bih1 = (const float*)d_in[9];
  const float* bhh1 = (const float*)d_in[10];
  const float* Whp  = (const float*)d_in[11];
  const float* bhp  = (const float*)d_in[12];
  float* out = (float*)d_out;

  dim3 grid(BDIM / 16);   // 512 workgroups, 16 batch rows each
  dim3 block(256);        // 8 wave32s on one WGP
  decoder_lstm_kernel<<<grid, block, 0, stream>>>(obs, h0g, c0g, Wih0, Whh0, bih0, bhh0,
                                                  Wih1, Whh1, bih1, bhh1, Whp, bhp, out);
}
